// SkeletonRecallLoss2D_48868137894335
// MI455X (gfx1250) — compile-verified, matrix operand order checked by reading, not run on hardware
//
#include <hip/hip_runtime.h>
#include <hip/hip_bf16.h>

typedef __attribute__((ext_vector_type(16))) _Float16 v16h;
typedef __attribute__((ext_vector_type(8)))  float    v8f;

#define IMG_H   512
#define IMG_W   512
#define WPR     16              // 32-bit words per row
#define WORDS   8192            // words per image (512*16)
#define NPIX    262144          // pixels per image
#define BATCH   32
#define MAX_ITERS 256
#define BPI     16              // reduction blocks per image

// ---------------- bit-parallel helpers ----------------

__device__ __forceinline__ unsigned ldw(const unsigned* buf, int r, int w) {
    if ((unsigned)r >= IMG_H || (unsigned)w >= WPR) return 0u;
    return buf[(r << 4) + w];
}

// full adder: s = a+b+c (sum bit), cy = carry bit
__device__ __forceinline__ void fa3(unsigned a, unsigned b, unsigned c,
                                    unsigned& s, unsigned& cy) {
    unsigned x = a ^ b;
    s  = x ^ c;
    cy = (a & b) | (c & x);
}

// per-bit popcount of 8 one-bit operands -> 4-bit count {b3,b2,b1,b0}
__device__ __forceinline__ void count8(unsigned v0, unsigned v1, unsigned v2, unsigned v3,
                                       unsigned v4, unsigned v5, unsigned v6, unsigned v7,
                                       unsigned& b0, unsigned& b1, unsigned& b2, unsigned& b3) {
    unsigned s1, c1, s2, c2, s3, c3;
    fa3(v0, v1, v2, s1, c1);
    fa3(v3, v4, v5, s2, c2);
    fa3(v6, v7, s1, s3, c3);
    unsigned k1 = s2 & s3;  b0 = s2 ^ s3;          // weight-1 -> b0, carry k1 (w2)
    unsigned t, u;          fa3(c1, c2, c3, t, u); // w2 bits -> t (w2), u (w4)
    b1 = t ^ k1;  unsigned k2 = t & k1;            // w2 -> b1, carry k2 (w4)
    b2 = u ^ k2;  b3 = u & k2;                     // w4 -> b2, carry b3 (w8)
}

// one Zhang-Suen substep on one 32-pixel word; returns new word
__device__ __forceinline__ unsigned thin_word(const unsigned* cur, int r, int w, bool first) {
    unsigned c = ldw(cur, r, w);
    if (c == 0u) return 0u;                        // removals only clear set bits
    unsigned uL = ldw(cur, r - 1, w - 1), uC = ldw(cur, r - 1, w), uR = ldw(cur, r - 1, w + 1);
    unsigned cL = ldw(cur, r,     w - 1),                          cR = ldw(cur, r,     w + 1);
    unsigned dL = ldw(cur, r + 1, w - 1), dC = ldw(cur, r + 1, w), dR = ldw(cur, r + 1, w + 1);
    // bit i == pixel x = w*32+i (LSB-first, matches ballot lane order)
    unsigned P2 = uC;                              // N
    unsigned P3 = (uC >> 1) | (uR << 31);          // NE  (east shift)
    unsigned P4 = (c  >> 1) | (cR << 31);          // E
    unsigned P5 = (dC >> 1) | (dR << 31);          // SE
    unsigned P6 = dC;                              // S
    unsigned P7 = (dC << 1) | (dL >> 31);          // SW  (west shift)
    unsigned P8 = (c  << 1) | (cL >> 31);          // W
    unsigned P9 = (uC << 1) | (uL >> 31);          // NW

    unsigned n0, n1, n2, n3;
    count8(P2, P3, P4, P5, P6, P7, P8, P9, n0, n1, n2, n3);
    unsigned ge2 = n1 | n2 | n3;                   // Bn >= 2
    unsigned le6 = ~(n3 | (n2 & n1 & n0));         // Bn <= 6

    unsigned t0 = ~P2 & P3, t1 = ~P3 & P4, t2 = ~P4 & P5, t3 = ~P5 & P6;
    unsigned t4 = ~P6 & P7, t5 = ~P7 & P8, t6 = ~P8 & P9, t7 = ~P9 & P2;
    unsigned a0, a1, a2, a3;
    count8(t0, t1, t2, t3, t4, t5, t6, t7, a0, a1, a2, a3);
    unsigned aeq1 = a0 & ~a1 & ~a2 & ~a3;          // A == 1

    unsigned c1m, c2m;
    if (first) { c1m = ~(P2 & P4 & P6); c2m = ~(P4 & P6 & P8); }
    else       { c1m = ~(P2 & P4 & P8); c2m = ~(P2 & P6 & P8); }

    unsigned remove = c & ge2 & le6 & aeq1 & c1m & c2m;
    return c & ~remove;
}

// fast sigmoid: v_exp_f32 + v_rcp_f32 (both TRANS32, co-execute with VALU/WMMA)
__device__ __forceinline__ float fsigmoid(float x) {
    return __builtin_amdgcn_rcpf(1.0f + __expf(-x));
}

// ---------------- kernel A: skeletonize + dilate (1 workgroup per image, all in LDS) ----------------

__global__ void skel_kernel(const float* __restrict__ mask,
                            unsigned* __restrict__ finalBits,
                            float* __restrict__ gtSum) {
    extern __shared__ unsigned lds[];              // [0,8192)=cur  [8192,16384)=nxt  [16384]=flag [16385]=cnt
    unsigned* cur = lds;
    unsigned* nxt = lds + WORDS;
    const int img  = blockIdx.x;
    const int tid  = threadIdx.x;
    const int lane = tid & 31;
    const float* m = mask + (size_t)img * NPIX;

    if (tid == 0) { lds[2 * WORDS] = 0u; lds[2 * WORDS + 1] = 0u; }

    // binarize -> bitmask via wave32 ballot (coalesced 128B loads per wave)
    for (int p = tid; p < NPIX; p += 1024) {
        unsigned b = __builtin_amdgcn_ballot_w32(m[p] > 0.0f);
        if (lane == 0) cur[p >> 5] = b;
    }
    __syncthreads();

    // iterative Zhang-Suen thinning, entirely in LDS
    for (int it = 0; it < MAX_ITERS; ++it) {
        unsigned diff = 0u;
        for (int wl = tid; wl < WORDS; wl += 1024) {          // substep 1: cur -> nxt
            int r = wl >> 4, w = wl & 15;
            unsigned c  = cur[wl];
            unsigned nw = thin_word(cur, r, w, true);
            nxt[wl] = nw;
            diff |= c ^ nw;
        }
        __syncthreads();
        for (int wl = tid; wl < WORDS; wl += 1024) {          // substep 2: nxt -> cur
            int r = wl >> 4, w = wl & 15;
            unsigned c  = nxt[wl];
            unsigned nw = thin_word(nxt, r, w, false);
            cur[wl] = nw;
            diff |= c ^ nw;
        }
        if (diff) atomicOr(&lds[2 * WORDS], 1u);
        __syncthreads();
        unsigned ch = lds[2 * WORDS];
        __syncthreads();                                      // all reads done before reset
        if (tid == 0) lds[2 * WORDS] = 0u;                    // next write (atomicOr) is after a barrier
        if (!ch) break;
    }
    __syncthreads();

    // dilate with disk radius 2 (13-tap OR), cur -> nxt
    for (int wl = tid; wl < WORDS; wl += 1024) {
        int r = wl >> 4, w = wl & 15;
        unsigned c  = ldw(cur, r, w), cL = ldw(cur, r, w - 1), cR = ldw(cur, r, w + 1);
        unsigned h2 = c | ((c >> 1) | (cR << 31)) | ((c << 1) | (cL >> 31))
                        | ((c >> 2) | (cR << 30)) | ((c << 2) | (cL >> 30));
        unsigned uC = ldw(cur, r - 1, w), uL = ldw(cur, r - 1, w - 1), uR = ldw(cur, r - 1, w + 1);
        unsigned h1u = uC | ((uC >> 1) | (uR << 31)) | ((uC << 1) | (uL >> 31));
        unsigned dC = ldw(cur, r + 1, w), dL = ldw(cur, r + 1, w - 1), dR = ldw(cur, r + 1, w + 1);
        unsigned h1d = dC | ((dC >> 1) | (dR << 31)) | ((dC << 1) | (dL >> 31));
        nxt[wl] = h2 | h1u | h1d | ldw(cur, r - 2, w) | ldw(cur, r + 2, w);
    }
    __syncthreads();

    // final = dilated & binary(mask); store bitmask + popcount (integer atomics only -> deterministic)
    unsigned cnt = 0u;
    unsigned* outw = finalBits + (size_t)img * WORDS;
    for (int p = tid; p < NPIX; p += 1024) {
        unsigned b = __builtin_amdgcn_ballot_w32(m[p] > 0.0f);
        if (lane == 0) {
            unsigned fw = b & nxt[p >> 5];
            outw[p >> 5] = fw;
            cnt += (unsigned)__popc(fw);
        }
    }
    if (lane == 0 && cnt) atomicAdd(&lds[2 * WORDS + 1], cnt);
    __syncthreads();
    if (tid == 0) gtSum[img] = (float)lds[2 * WORDS + 1];
}

// ---------------- kernel B: masked-sigmoid reduction via WMMA (ones-matrix rowsum trick) ----------------

__global__ void inter_kernel(const float* __restrict__ logits,
                             const unsigned* __restrict__ finalBits,
                             float* __restrict__ partials) {
    const int img  = blockIdx.x / BPI;
    const int blk  = blockIdx.x % BPI;
    const int tid  = threadIdx.x;                  // 256 threads = 8 waves
    const int wv   = tid >> 5;
    const int lane = tid & 31;
    const float*    lg = logits    + (size_t)img * NPIX;
    const unsigned* fb = finalBits + (size_t)img * WORDS;

    __shared__ float waveSum[8];

    v16h ones;
#pragma unroll
    for (int i = 0; i < 16; ++i) ones[i] = (_Float16)1.0f;

    v8f acc = {0.f, 0.f, 0.f, 0.f, 0.f, 0.f, 0.f, 0.f};

    const int waveBase = blk * (NPIX / BPI) + wv * 2048;      // 2048 pixels per wave
#pragma unroll
    for (int rd = 0; rd < 4; ++rd) {                          // 512 pixels per WMMA round
        const int base = waveBase + rd * 512;
        v16h a;
#pragma unroll
        for (int j = 0; j < 4; ++j) {
            const int p0 = base + j * 128 + lane * 4;         // coalesced float4 per lane
            if (j == 0) __builtin_prefetch(lg + p0 + 2048, 0, 0);   // global_prefetch_b8
            const float4 v = *(const float4*)(lg + p0);
            const unsigned bits = (fb[p0 >> 5] >> (p0 & 31)) & 0xFu;
            float s0 = fsigmoid(v.x);
            float s1 = fsigmoid(v.y);
            float s2 = fsigmoid(v.z);
            float s3 = fsigmoid(v.w);
            a[j * 4 + 0] = (bits & 1u) ? (_Float16)s0 : (_Float16)0.0f;
            a[j * 4 + 1] = (bits & 2u) ? (_Float16)s1 : (_Float16)0.0f;
            a[j * 4 + 2] = (bits & 4u) ? (_Float16)s2 : (_Float16)0.0f;
            a[j * 4 + 3] = (bits & 8u) ? (_Float16)s3 : (_Float16)0.0f;
        }
        // D = A x Ones + C : every column of D holds rowsums, so Sum(D) = 16 * Sum(A)
        acc = __builtin_amdgcn_wmma_f32_16x16x32_f16(
            /*neg_a=*/false, a, /*neg_b=*/false, ones,
            /*c_mod=*/(short)0, acc, /*reuse_a=*/false, /*reuse_b=*/false);
    }

    float part = acc[0] + acc[1] + acc[2] + acc[3] + acc[4] + acc[5] + acc[6] + acc[7];
#pragma unroll
    for (int off = 16; off; off >>= 1) part += __shfl_xor(part, off, 32);
    if (lane == 0) waveSum[wv] = part;
    __syncthreads();
    if (tid == 0) {
        float s = 0.0f;
        for (int i = 0; i < 8; ++i) s += waveSum[i];          // fixed order -> deterministic
        partials[img * BPI + blk] = s * (1.0f / 16.0f);       // undo 16x column broadcast
    }
}

// ---------------- kernel C: per-image recall + mean ----------------

__global__ void finalize_kernel(const float* __restrict__ gtSum,
                                const float* __restrict__ partials,
                                float* __restrict__ out) {
    const int b = threadIdx.x;                     // 32 threads = 1 wave
    float inter = 0.0f;
    for (int i = 0; i < BPI; ++i) inter += partials[b * BPI + i];
    float t = 1.0f - (inter + 1e-6f) / (gtSum[b] + 1e-6f);
#pragma unroll
    for (int off = 16; off; off >>= 1) t += __shfl_xor(t, off, 32);
    if (b == 0) out[0] = t * (1.0f / 32.0f);
}

// ---------------- launch ----------------

extern "C" void kernel_launch(void* const* d_in, const int* in_sizes, int n_in,
                              void* d_out, int out_size, void* d_ws, size_t ws_size,
                              hipStream_t stream) {
    const float* logits = (const float*)d_in[0];
    const float* gtmask = (const float*)d_in[1];

    unsigned* finalBits = (unsigned*)d_ws;                              // 32 * 8192 u32 = 1 MB
    float*    gtSum     = (float*)((char*)d_ws + (size_t)BATCH * WORDS * 4);   // 32 f32
    float*    partials  = gtSum + BATCH;                                // 32*16 f32

    // A: in-LDS bit-parallel thinning + dilation. Dynamic LDS: 2 bit-buffers + flag + counter.
    const size_t ldsBytes = (size_t)(2 * WORDS + 2) * sizeof(unsigned);
    skel_kernel<<<BATCH, 1024, ldsBytes, stream>>>(gtmask, finalBits, gtSum);

    // B: WMMA masked-sigmoid reduction
    inter_kernel<<<BATCH * BPI, 256, 0, stream>>>(logits, finalBits, partials);

    // C: scalar loss
    finalize_kernel<<<1, 32, 0, stream>>>(gtSum, partials, (float*)d_out);
}
